// CrossAttentionLayer_59785944760594
// MI455X (gfx1250) — matrix operand order repaired and checked by reading, not compiled, not run on hardware
//
#include <hip/hip_runtime.h>
#include <hip/hip_bf16.h>

typedef __bf16 bf16;
typedef __attribute__((ext_vector_type(8)))  __bf16 v8bf;
typedef __attribute__((ext_vector_type(16))) __bf16 v16bf;
typedef __attribute__((ext_vector_type(8)))  float  v8f;
typedef __attribute__((ext_vector_type(4)))  int    v4i;

#define TILE 128
#define KT   32

#define GAS __attribute__((address_space(1)))
#define LAS __attribute__((address_space(3)))

// Async global->LDS staging (CDNA5 GLOBAL_LOAD_ASYNC_TO_LDS_B128, ASYNCcnt).
// Builtin signature (from hipcc diagnostics): (int4 AS1*, int4 AS3*, imm, imm).
#if defined(__AMDGCN__) && __has_builtin(__builtin_amdgcn_global_load_async_to_lds_b128)
#define HAVE_ASYNC 1
#define ASYNC_CP16(lds, g)                                                   \
    __builtin_amdgcn_global_load_async_to_lds_b128(                          \
        (GAS v4i*)(void*)(g), (LAS v4i*)(void*)(lds), 0, 0)
#if __has_builtin(__builtin_amdgcn_s_wait_asynccnt)
#define ASYNC_WAIT() __builtin_amdgcn_s_wait_asynccnt(0)
#else
#define ASYNC_WAIT() asm volatile("s_wait_asynccnt 0" ::: "memory")
#endif
#else
#define HAVE_ASYNC 0
#define ASYNC_WAIT() ((void)0)
#endif

static __device__ __forceinline__ v8bf cvt8(const float* s) {
    float4 x = ((const float4*)s)[0];
    float4 y = ((const float4*)s)[1];
    v8bf r;
    r[0] = (bf16)x.x; r[1] = (bf16)x.y; r[2] = (bf16)x.z; r[3] = (bf16)x.w;
    r[4] = (bf16)y.x; r[5] = (bf16)y.y; r[6] = (bf16)y.z; r[7] = (bf16)y.w;
    return r;
}

// C[row, col] = sum_k A[row, k] * B'[k, col]
//  - IN_F32 : A and B sources are f32, converted to bf16 while staging
//  - B_NT   : B source is row-major (cols x Kdim); else (Kdim x cols) -> LDS transpose
//  - OUT_F32: C written as f32, else bf16
// 256 threads = 8 waves; block tile 128x128; wave tile 32x64 (4x2 wave grid,
// 2 A-frags x 4 B-frags -> 8 WMMAs per 12 b128 LDS loads).
// Double-buffered LDS; bf16 sources staged with async-to-LDS when available.
template<bool IN_F32, bool B_NT, bool OUT_F32>
__global__ __launch_bounds__(256)
void gemm128_wmma(const void* __restrict__ Ap, long long lda, long long strideA,
                  const void* __restrict__ Bp, long long ldb, long long strideB,
                  void*       __restrict__ Cp, long long ldc, long long strideC,
                  int Kdim)
{
    __shared__ __align__(16) bf16 At[2][TILE][KT];
    __shared__ __align__(16) bf16 Bt[2][TILE][KT];

    const int tid  = threadIdx.x;
    const int lane = tid & 31;
    const int wv   = tid >> 5;
    const int rg   = wv >> 1;          // wave row-group: 0..3 (32 rows each)
    const int cg   = wv & 1;           // wave col-group: 0..1 (64 cols each)
    const long long rowbase = (long long)blockIdx.y * TILE;
    const long long colbase = (long long)blockIdx.x * TILE;
    const int batch = blockIdx.z;

    const float* Af = (const float*)Ap + (long long)batch * strideA;
    const bf16*  Ab = (const bf16*) Ap + (long long)batch * strideA;
    const float* Bf = (const float*)Bp + (long long)batch * strideB;
    const bf16*  Bb = (const bf16*) Bp + (long long)batch * strideB;

    // Per-thread staging geometry: 16 contiguous elements each.
    const int ar  = tid >> 1;          // A / B-NT: row (or col), half-row of 16
    const int akh = (tid & 1) * 16;    //           k offset within row
    const int vk  = tid >> 3;          // B-NN: k row
    const int vc  = (tid & 7) * 16;    //        col offset (16 contiguous cols)

    v8bf ra0, ra1, rb0, rb1;           // register staging (sync paths)
    v8f acc[8] = {};                   // acc[i*4+t]: rows i*16, cols t*16

    auto load_phase = [&](int k0, int db) {
        // ---- A tile ----
        if (IN_F32) {
            const float* s = &Af[(rowbase + ar) * lda + (k0 + akh)];
            ra0 = cvt8(s);
            ra1 = cvt8(s + 8);
        } else {
            const bf16* g = &Ab[(rowbase + ar) * lda + (k0 + akh)];
#if HAVE_ASYNC
            ASYNC_CP16(&At[db][ar][akh],     g);
            ASYNC_CP16(&At[db][ar][akh + 8], g + 8);
#else
            ra0 = ((const v8bf*)g)[0];
            ra1 = ((const v8bf*)g)[1];
#endif
        }
        // ---- B tile ----
        if (B_NT) {
            if (IN_F32) {
                const float* s = &Bf[(colbase + ar) * ldb + (k0 + akh)];
                rb0 = cvt8(s);
                rb1 = cvt8(s + 8);
            } else {
                const bf16* g = &Bb[(colbase + ar) * ldb + (k0 + akh)];
#if HAVE_ASYNC
                ASYNC_CP16(&Bt[db][ar][akh],     g);
                ASYNC_CP16(&Bt[db][ar][akh + 8], g + 8);
#else
                rb0 = ((const v8bf*)g)[0];
                rb1 = ((const v8bf*)g)[1];
#endif
            }
        } else {
            // (Kdim x cols): coalesced 16 contiguous cols, transposed on store.
            const bf16* g = &Bb[(long long)(k0 + vk) * ldb + (colbase + vc)];
            rb0 = ((const v8bf*)g)[0];
            rb1 = ((const v8bf*)g)[1];
        }
    };

    auto store_phase = [&](int db) {
        if (IN_F32 || !HAVE_ASYNC) {
            ((v8bf*)&At[db][ar][akh])[0] = ra0;
            ((v8bf*)&At[db][ar][akh])[1] = ra1;
        }
        if (B_NT) {
            if (IN_F32 || !HAVE_ASYNC) {
                ((v8bf*)&Bt[db][ar][akh])[0] = rb0;
                ((v8bf*)&Bt[db][ar][akh])[1] = rb1;
            }
        } else {
            #pragma unroll
            for (int j = 0; j < 8; ++j) Bt[db][vc + j][vk]     = rb0[j];
            #pragma unroll
            for (int j = 0; j < 8; ++j) Bt[db][vc + 8 + j][vk] = rb1[j];
        }
    };

    auto compute = [&](int db) {
        // A fragments 16x32: M = lane&15; this code splits K into contiguous halves by lane>=16.
        const int lanelo = lane & 15;
        const int ksel   = (lane >> 4) << 4;
        v16bf af[2];
        #pragma unroll
        for (int i = 0; i < 2; ++i) {
            const int am = rg * 32 + i * 16 + lanelo;
            v8bf alo = *(const v8bf*)&At[db][am][ksel];
            v8bf ahi = *(const v8bf*)&At[db][am][ksel + 8];
            af[i] = __builtin_shufflevector(alo, ahi,
                0,1,2,3,4,5,6,7,8,9,10,11,12,13,14,15);
        }
        // B fragments 32x16: lane = col, K half by lane>=16; each reused twice.
        const int bks = (lane >> 4) << 4;
        #pragma unroll
        for (int t = 0; t < 4; ++t) {
            v16bf bfrag = *(const v16bf*)&Bt[db][cg * 64 + t * 16 + lanelo][bks];
            #pragma unroll
            for (int i = 0; i < 2; ++i) {
                acc[i * 4 + t] = __builtin_amdgcn_wmma_f32_16x16x32_bf16(
                    false, af[i], false, bfrag, (short)0, acc[i * 4 + t],
                    false, false);
            }
        }
    };

    const int nsteps = Kdim / KT;

    // Prologue: stage step 0 into buffer 0.
    load_phase(0, 0);
    store_phase(0);
    ASYNC_WAIT();
    __syncthreads();

    int buf = 0;
    for (int s = 0; s < nsteps; ++s) {
        if (s + 1 < nsteps) load_phase((s + 1) * KT, buf ^ 1);  // prefetch next
        compute(buf);
        if (s + 1 < nsteps) store_phase(buf ^ 1);
        ASYNC_WAIT();
        __syncthreads();
        buf ^= 1;
    }

    // Store C: VGPR r -> M = r + 8*(lane>=16), N = lane&15 (ISA 7.12.2).
    float* Cf = (float*)Cp + (long long)batch * strideC;
    bf16*  Cb = (bf16*) Cp + (long long)batch * strideC;
    const int rsel = (lane >> 4) << 3;
    #pragma unroll
    for (int i = 0; i < 2; ++i) {
        #pragma unroll
        for (int t = 0; t < 4; ++t) {
            #pragma unroll
            for (int r = 0; r < 8; ++r) {
                long long row = rowbase + rg * 32 + i * 16 + r + rsel;
                long long col = colbase + cg * 64 + t * 16 + (lane & 15);
                if (OUT_F32) Cf[row * ldc + col] = acc[i * 4 + t][r];
                else         Cb[row * ldc + col] = (bf16)acc[i * 4 + t][r];
            }
        }
    }
}

// P = softmax(where(mask==0, -1e10, E) * scale) per row of 2048, bf16 out.
__global__ __launch_bounds__(256)
void softmax_mask_kernel(const float* __restrict__ E, const int* __restrict__ mask,
                         bf16* __restrict__ P, float scale)
{
    const long long row = blockIdx.x;
    const float* e  = E    + row * 2048;
    const int*   mk = mask + row * 2048;
    bf16*        p  = P    + row * 2048;
    const int tid  = threadIdx.x;
    const int lane = tid & 31;
    const int wv   = tid >> 5;
    const int base = tid * 8;
    __shared__ float redmax[8];
    __shared__ float redsum[8];

    float4 e0 = ((const float4*)(e + base))[0];
    float4 e1 = ((const float4*)(e + base))[1];
    int4   m0 = ((const int4*)(mk + base))[0];
    int4   m1 = ((const int4*)(mk + base))[1];

    float v[8];
    v[0] = (m0.x == 0 ? -1e10f : e0.x) * scale;
    v[1] = (m0.y == 0 ? -1e10f : e0.y) * scale;
    v[2] = (m0.z == 0 ? -1e10f : e0.z) * scale;
    v[3] = (m0.w == 0 ? -1e10f : e0.w) * scale;
    v[4] = (m1.x == 0 ? -1e10f : e1.x) * scale;
    v[5] = (m1.y == 0 ? -1e10f : e1.y) * scale;
    v[6] = (m1.z == 0 ? -1e10f : e1.z) * scale;
    v[7] = (m1.w == 0 ? -1e10f : e1.w) * scale;

    float lmax = v[0];
    #pragma unroll
    for (int i = 1; i < 8; ++i) lmax = fmaxf(lmax, v[i]);
    #pragma unroll
    for (int o = 16; o > 0; o >>= 1) lmax = fmaxf(lmax, __shfl_xor(lmax, o, 32));
    if (lane == 0) redmax[wv] = lmax;
    __syncthreads();
    float m = redmax[0];
    #pragma unroll
    for (int w = 1; w < 8; ++w) m = fmaxf(m, redmax[w]);

    float lsum = 0.f;
    #pragma unroll
    for (int i = 0; i < 8; ++i) { v[i] = __expf(v[i] - m); lsum += v[i]; }
    #pragma unroll
    for (int o = 16; o > 0; o >>= 1) lsum += __shfl_xor(lsum, o, 32);
    if (lane == 0) redsum[wv] = lsum;
    __syncthreads();
    float tot = 0.f;
    #pragma unroll
    for (int w = 0; w < 8; ++w) tot += redsum[w];
    float inv = 1.0f / tot;

    v8bf out;
    #pragma unroll
    for (int i = 0; i < 8; ++i) out[i] = (bf16)(v[i] * inv);
    *(v8bf*)(p + base) = out;
}

extern "C" void kernel_launch(void* const* d_in, const int* in_sizes, int n_in,
                              void* d_out, int out_size, void* d_ws, size_t ws_size,
                              hipStream_t stream) {
    constexpr int B = 8, N = 2048, M = 2048, D = 1024, HID = 1024, OUT = 1024;

    const float* h1   = (const float*)d_in[0];   // (B,N,D)
    const float* h2   = (const float*)d_in[1];   // (B,M,D)
    const int*   mask = (const int*)  d_in[2];   // (B,N,M)
    const float* Wq   = (const float*)d_in[3];   // (HID,D)
    const float* Wk   = (const float*)d_in[4];   // (HID,D)
    const float* Wv   = (const float*)d_in[5];   // (OUT,D)
    float*       out  = (float*)d_out;           // (B,N,OUT)

    // Workspace: Q(32MB) K(32MB) V(32MB) bf16, E f32 (128MB), P bf16 (64MB).
    bf16*  Qb = (bf16*)d_ws;
    bf16*  Kb = Qb + (size_t)B * N * HID;
    bf16*  Vb = Kb + (size_t)B * M * HID;
    float* E  = (float*)(Vb + (size_t)B * M * OUT);
    bf16*  P  = (bf16*)(E + (size_t)B * N * M);

    dim3 blk(256);

    // Projections (weights shared across batch; rows flattened).
    gemm128_wmma<true, true, false><<<dim3(HID / TILE, (B * N) / TILE, 1), blk, 0, stream>>>(
        h1, D, 0, Wq, D, 0, Qb, HID, 0, D);
    gemm128_wmma<true, true, false><<<dim3(HID / TILE, (B * M) / TILE, 1), blk, 0, stream>>>(
        h2, D, 0, Wk, D, 0, Kb, HID, 0, D);
    gemm128_wmma<true, true, false><<<dim3(OUT / TILE, (B * M) / TILE, 1), blk, 0, stream>>>(
        h2, D, 0, Wv, D, 0, Vb, OUT, 0, D);

    // Scores: E[b] = Q[b] x K[b]^T, batched, f32 out.
    gemm128_wmma<false, true, true><<<dim3(M / TILE, N / TILE, B), blk, 0, stream>>>(
        Qb, HID, (long long)N * HID,
        Kb, HID, (long long)M * HID,
        E,  M,   (long long)N * M,
        HID);

    // Masked, scaled softmax -> bf16 probabilities.
    softmax_mask_kernel<<<B * N, blk, 0, stream>>>(E, mask, P, 1.0f / 32.0f);

    // Output: out[b] = P[b] x V[b] (V staged NN / transposed in LDS).
    gemm128_wmma<false, false, true><<<dim3(OUT / TILE, N / TILE, B), blk, 0, stream>>>(
        P,  M,   (long long)N * M,
        Vb, OUT, (long long)M * OUT,
        out, OUT, (long long)N * OUT,
        M);
}
